// MambaLayer_7859790152262
// MI455X (gfx1250) — compile-verified, matrix operand order checked by reading, not run on hardware
//
#include <hip/hip_runtime.h>
#include <hip/hip_bf16.h>

typedef _Float16 half_t;
typedef __attribute__((ext_vector_type(16))) _Float16 v16h;
typedef __attribute__((ext_vector_type(8)))  _Float16 v8h;
typedef __attribute__((ext_vector_type(8)))  float    v8f;

#define DIMV     256
#define D_INNER  512
#define D_STATE  64
#define DT_RANK  16
#define B_SZ     2
#define L_SEQ    1024
#define M_ROWS   (B_SZ * L_SEQ)          // 2048
#define N_XZ     (2 * D_INNER)           // 1024
#define N_DBL    (DT_RANK + 2 * D_STATE) // 144

// ---------------------------------------------------------------- utilities
__device__ __forceinline__ float wave_reduce_add(float v) {
#pragma unroll
    for (int off = 16; off > 0; off >>= 1)
        v += __shfl_xor(v, off, 32);
    return v;
}

union frag16 { v16h v; v8h h[2]; };

__device__ __forceinline__ v8f wmma_f16(const frag16& a, const frag16& b, v8f c) {
    return __builtin_amdgcn_wmma_f32_16x16x32_f16(
               false, a.v, false, b.v, (short)0, c, false, false);
}

__global__ void k_f32_to_f16(const float* __restrict__ in, half_t* __restrict__ out, int n) {
    int i = blockIdx.x * blockDim.x + threadIdx.x;
    if (i < n) out[i] = (half_t)in[i];
}

// ------------------------------------------------- LayerNorm (wave per row)
__global__ void k_layernorm(const float* __restrict__ x,
                            const float* __restrict__ g,
                            const float* __restrict__ bta,
                            half_t* __restrict__ out) {
    int row  = blockIdx.x * (blockDim.x >> 5) + (threadIdx.x >> 5);
    int lane = threadIdx.x & 31;
    const float* xr = x + (size_t)row * DIMV;
    float vals[8];
    float s = 0.f, s2 = 0.f;
#pragma unroll
    for (int i = 0; i < 8; ++i) {
        float v = xr[lane + 32 * i];
        vals[i] = v; s += v; s2 += v * v;
    }
    s  = wave_reduce_add(s);
    s2 = wave_reduce_add(s2);
    float mu  = s * (1.f / DIMV);
    float var = s2 * (1.f / DIMV) - mu * mu;
    float rs  = rsqrtf(var + 1e-5f);
    half_t* orow = out + (size_t)row * DIMV;
#pragma unroll
    for (int i = 0; i < 8; ++i) {
        int c = lane + 32 * i;
        orow[c] = (half_t)((vals[i] - mu) * rs * g[c] + bta[c]);
    }
}

// --------------- generic WMMA GEMM (16x16/wave): C[M,N] = A[M,K] * W[N,K]^T
__global__ void k_wmma_gemm16(const half_t* __restrict__ A,
                              const half_t* __restrict__ W,
                              float* __restrict__ C,
                              int M, int N, int K) {
    int wave = (blockIdx.x * blockDim.x + threadIdx.x) >> 5;
    int lane = threadIdx.x & 31;
    int ntiles = N >> 4;
    int total  = (M >> 4) * ntiles;
    if (wave >= total) return;          // wave-uniform: EXEC stays all-ones
    int tM = wave / ntiles;
    int tN = wave % ntiles;

    int g = (lane >> 4) * 8;            // per ISA 16-bit A 16x32 lane layout
    const half_t* arow = A + (size_t)(tM * 16 + (lane & 15)) * K + g;
    const half_t* wrow = W + (size_t)(tN * 16 + (lane & 15)) * K + g;

    v8f acc = {};
    for (int kb = 0; kb < K; kb += 32) {
        __builtin_prefetch(arow + kb + 32, 0, 3);   // near-scope prefetch
        __builtin_prefetch(wrow + kb + 32, 0, 3);
        frag16 af, bf;
        af.h[0] = *(const v8h*)(arow + kb);
        af.h[1] = *(const v8h*)(arow + kb + 16);
        bf.h[0] = *(const v8h*)(wrow + kb);
        bf.h[1] = *(const v8h*)(wrow + kb + 16);
        acc = wmma_f16(af, bf, acc);
    }

    int mb = tM * 16 + (lane >> 4) * 8;
    float* crow = C + (size_t)mb * N + tN * 16 + (lane & 15);
#pragma unroll
    for (int r = 0; r < 8; ++r) crow[(size_t)r * N] = acc[r];
}

// ------- blocked WMMA GEMM (32x32/wave): 2x register reuse of fragments.
// Requires M % 32 == 0 and N % 32 == 0.
__global__ void k_wmma_gemm32(const half_t* __restrict__ A,
                              const half_t* __restrict__ W,
                              float* __restrict__ C,
                              int M, int N, int K) {
    int wave = (blockIdx.x * blockDim.x + threadIdx.x) >> 5;
    int lane = threadIdx.x & 31;
    int nst   = N >> 5;
    int total = (M >> 5) * nst;
    if (wave >= total) return;          // wave-uniform
    int sM = wave / nst;
    int sN = wave % nst;

    int g = (lane >> 4) * 8;
    int r16 = lane & 15;
    const half_t* a0 = A + (size_t)(sM * 32 + r16) * K + g;
    const half_t* a1 = a0 + (size_t)16 * K;
    const half_t* w0 = W + (size_t)(sN * 32 + r16) * K + g;
    const half_t* w1 = w0 + (size_t)16 * K;

    v8f acc00 = {}, acc01 = {}, acc10 = {}, acc11 = {};
    for (int kb = 0; kb < K; kb += 32) {
        __builtin_prefetch(a0 + kb + 32, 0, 3);
        __builtin_prefetch(a1 + kb + 32, 0, 3);
        __builtin_prefetch(w0 + kb + 32, 0, 3);
        __builtin_prefetch(w1 + kb + 32, 0, 3);
        frag16 af0, af1, bf0, bf1;
        af0.h[0] = *(const v8h*)(a0 + kb);  af0.h[1] = *(const v8h*)(a0 + kb + 16);
        af1.h[0] = *(const v8h*)(a1 + kb);  af1.h[1] = *(const v8h*)(a1 + kb + 16);
        bf0.h[0] = *(const v8h*)(w0 + kb);  bf0.h[1] = *(const v8h*)(w0 + kb + 16);
        bf1.h[0] = *(const v8h*)(w1 + kb);  bf1.h[1] = *(const v8h*)(w1 + kb + 16);
        acc00 = wmma_f16(af0, bf0, acc00);
        acc01 = wmma_f16(af0, bf1, acc01);
        acc10 = wmma_f16(af1, bf0, acc10);
        acc11 = wmma_f16(af1, bf1, acc11);
    }

    int mb = (lane >> 4) * 8;           // row offset within a 16-tile
    int n  = lane & 15;                 // col within a 16-tile
    float* cbase = C + (size_t)(sM * 32) * N + sN * 32;
#pragma unroll
    for (int r = 0; r < 8; ++r) {
        cbase[(size_t)(mb + r)      * N + n]      = acc00[r];
        cbase[(size_t)(mb + r)      * N + 16 + n] = acc01[r];
        cbase[(size_t)(16 + mb + r) * N + n]      = acc10[r];
        cbase[(size_t)(16 + mb + r) * N + 16 + n] = acc11[r];
    }
}

// -------------------- causal depthwise conv (k=2) + bias + SiLU -> u, u_h
__global__ void k_conv_silu(const float* __restrict__ xz,
                            const float* __restrict__ wconv,
                            const float* __restrict__ bconv,
                            float* __restrict__ u, half_t* __restrict__ u_h) {
    int idx = blockIdx.x * blockDim.x + threadIdx.x;      // m*512 + d
    if (idx >= M_ROWS * D_INNER) return;
    int d = idx & (D_INNER - 1);
    int m = idx >> 9;
    int l = m & (L_SEQ - 1);
    float cur  = xz[(size_t)m * N_XZ + d];
    float prev = (l > 0) ? xz[(size_t)(m - 1) * N_XZ + d] : 0.f;
    float v  = wconv[d * 2 + 0] * prev + wconv[d * 2 + 1] * cur + bconv[d];
    float su = v / (1.f + __expf(-v));
    u[idx]   = su;
    u_h[idx] = (half_t)su;
}

// ------------------------- dt = softplus(dt_low @ W_dt^T + b_dt)  (K = 16)
__global__ void k_dt(const float* __restrict__ dbl,
                     const float* __restrict__ Wdt,
                     const float* __restrict__ bdt,
                     float* __restrict__ dt) {
    int idx = blockIdx.x * blockDim.x + threadIdx.x;      // m*512 + d
    if (idx >= M_ROWS * D_INNER) return;
    int d = idx & (D_INNER - 1);
    int m = idx >> 9;
    const float* row = dbl + (size_t)m * N_DBL;
    const float* wd  = Wdt + d * DT_RANK;
    float acc = bdt[d];
#pragma unroll
    for (int r = 0; r < DT_RANK; ++r) acc += row[r] * wd[r];
    dt[idx] = (acc > 20.f) ? acc : log1pf(__expf(acc));
}

// ---------------- fused selective scan: wave per (b,d), 2 states per lane
__global__ void k_scan(const float* __restrict__ dt,
                       const float* __restrict__ u,
                       const float* __restrict__ dbl,
                       const float* __restrict__ xz,
                       const float* __restrict__ A_log,
                       const float* __restrict__ Dp,
                       half_t* __restrict__ y_h) {
    int w    = blockIdx.x * (blockDim.x >> 5) + (threadIdx.x >> 5); // 0..1023
    int lane = threadIdx.x & 31;
    int b = w >> 9;
    int d = w & (D_INNER - 1);
    int n0 = lane * 2;

    float A0 = -__expf(A_log[d * D_STATE + n0]);
    float A1 = -__expf(A_log[d * D_STATE + n0 + 1]);
    float Dv = Dp[d];
    float h0 = 0.f, h1 = 0.f;
    size_t mbase = (size_t)b * L_SEQ;

    for (int l = 0; l < L_SEQ; ++l) {
        size_t m   = mbase + l;
        float dtv  = dt[m * D_INNER + d];
        float uv   = u [m * D_INNER + d];
        const float* brow = dbl + m * N_DBL;
        float2 Bv = *(const float2*)(brow + DT_RANK + n0);
        float2 Cv = *(const float2*)(brow + DT_RANK + D_STATE + n0);
        float dtu = dtv * uv;
        h0 = __expf(dtv * A0) * h0 + dtu * Bv.x;
        h1 = __expf(dtv * A1) * h1 + dtu * Bv.y;
        float part = h0 * Cv.x + h1 * Cv.y;
        part = wave_reduce_add(part);
        if (lane == 0) {
            float yv = part + uv * Dv;
            float zv = xz[m * N_XZ + D_INNER + d];
            yv *= zv / (1.f + __expf(-zv));     // * silu(z)
            y_h[m * D_INNER + d] = (half_t)yv;
        }
    }
}

// ---------------------------------------------------------------- launcher
extern "C" void kernel_launch(void* const* d_in, const int* in_sizes, int n_in,
                              void* d_out, int out_size, void* d_ws, size_t ws_size,
                              hipStream_t stream) {
    const float* x      = (const float*)d_in[0];
    const float* ln_g   = (const float*)d_in[1];
    const float* ln_b   = (const float*)d_in[2];
    const float* W_in   = (const float*)d_in[3];
    const float* W_conv = (const float*)d_in[4];
    const float* b_conv = (const float*)d_in[5];
    const float* W_xp   = (const float*)d_in[6];
    const float* W_dt   = (const float*)d_in[7];
    const float* b_dt   = (const float*)d_in[8];
    const float* A_log  = (const float*)d_in[9];
    const float* Dp     = (const float*)d_in[10];
    const float* W_out  = (const float*)d_in[11];
    float* out = (float*)d_out;

    // workspace carve-out (256B aligned slabs)
    size_t off = 0;
    auto alloc = [&](size_t bytes) {
        void* p = (char*)d_ws + off;
        off += (bytes + 255) & ~(size_t)255;
        return p;
    };
    half_t* W_in_h  = (half_t*)alloc((size_t)N_XZ  * DIMV    * 2);
    half_t* W_xp_h  = (half_t*)alloc((size_t)N_DBL * D_INNER * 2);
    half_t* W_out_h = (half_t*)alloc((size_t)DIMV  * D_INNER * 2);
    half_t* xn_h    = (half_t*)alloc((size_t)M_ROWS * DIMV    * 2);
    float*  xzb     = (float*) alloc((size_t)M_ROWS * N_XZ    * 4);
    float*  u       = (float*) alloc((size_t)M_ROWS * D_INNER * 4);
    half_t* u_h     = (half_t*)alloc((size_t)M_ROWS * D_INNER * 2);
    float*  dbl     = (float*) alloc((size_t)M_ROWS * N_DBL   * 4);
    float*  dt      = (float*) alloc((size_t)M_ROWS * D_INNER * 4);
    half_t* y_h     = (half_t*)alloc((size_t)M_ROWS * D_INNER * 2);

    // 1) weights -> f16
    {
        int n1 = N_XZ * DIMV, n2 = N_DBL * D_INNER, n3 = DIMV * D_INNER;
        k_f32_to_f16<<<(n1 + 255) / 256, 256, 0, stream>>>(W_in,  W_in_h,  n1);
        k_f32_to_f16<<<(n2 + 255) / 256, 256, 0, stream>>>(W_xp,  W_xp_h,  n2);
        k_f32_to_f16<<<(n3 + 255) / 256, 256, 0, stream>>>(W_out, W_out_h, n3);
    }

    // 2) LayerNorm -> xn_h  (8 waves/block, wave per row)
    k_layernorm<<<M_ROWS / 8, 256, 0, stream>>>(x, ln_g, ln_b, xn_h);

    auto launch_gemm32 = [&](const half_t* A, const half_t* W, float* C,
                             int M, int N, int K) {
        int tiles  = (M / 32) * (N / 32);
        int blocks = (tiles + 7) / 8;               // 8 waves / 256-thread block
        k_wmma_gemm32<<<blocks, 256, 0, stream>>>(A, W, C, M, N, K);
    };

    // 3) xz = xn @ W_in^T        (2048 x 1024, K=256) — 32x32 blocked
    launch_gemm32(xn_h, W_in_h, xzb, M_ROWS, N_XZ, DIMV);

    // 4) u = silu(depthwise_conv(xi) + b_conv)
    k_conv_silu<<<(M_ROWS * D_INNER) / 256, 256, 0, stream>>>(xzb, W_conv, b_conv, u, u_h);

    // 5) dbl = u @ W_xp^T        (2048 x 144, K=512) — N=9x16 tiles, generic
    {
        int tiles  = (M_ROWS / 16) * (N_DBL / 16);
        int blocks = (tiles + 7) / 8;
        k_wmma_gemm16<<<blocks, 256, 0, stream>>>(u_h, W_xp_h, dbl,
                                                  M_ROWS, N_DBL, D_INNER);
    }

    // 6) dt = softplus(dt_low @ W_dt^T + b_dt)
    k_dt<<<(M_ROWS * D_INNER) / 256, 256, 0, stream>>>(dbl, W_dt, b_dt, dt);

    // 7) fused selective scan + skip + gate -> y_h (f16)
    k_scan<<<(B_SZ * D_INNER) / 8, 256, 0, stream>>>(dt, u, dbl, xzb, A_log, Dp, y_h);

    // 8) out = y @ W_out^T       (2048 x 256, K=512) — 32x32 blocked
    launch_gemm32(y_h, W_out_h, out, M_ROWS, DIMV, D_INNER);
}
